// AttentionHead_35785667510316
// MI455X (gfx1250) — matrix operand order repaired and checked by reading, not compiled, not run on hardware
//
#include <hip/hip_runtime.h>

// ---------------------------------------------------------------------------
// Causal MHA for MI455X (gfx1250, wave32, WMMA bf16 16x16x32, f32 accum)
//   B=4, T=2048, H=16, Dh=64, Dmodel=1024
// v5: exp2-domain online softmax (log2e folded into Q projection, bare
//     v_exp_f32), wave-private P-tile ordering via wave_barrier (1 block
//     barrier per attention chunk), async global->LDS DMA staging of GEMM tiles and K chunks.
// ---------------------------------------------------------------------------

typedef __attribute__((ext_vector_type(16))) __bf16 v16bf;
typedef __attribute__((ext_vector_type(8)))  float  v8f;

#define BATCH   4
#define SEQ     2048
#define NHEADS  16
#define DHEAD   64
#define DMODEL  1024
#define MTOT    (BATCH * SEQ)     // 8192 rows

// 1/sqrt(Dh) * log2(e): attention scores come out of WMMA in exp2 domain
#define QSCALE  (0.125f * 1.4426950408889634f)

template <bool B> struct BoolC { static constexpr bool value = B; };

__device__ __forceinline__ unsigned short f2bf(float f) {
    unsigned int u = __builtin_bit_cast(unsigned int, f);
    u += 0x7FFFu + ((u >> 16) & 1u);       // round-to-nearest-even
    return (unsigned short)(u >> 16);
}
__device__ __forceinline__ v8f v8f_zero() {
    v8f z;
#pragma unroll
    for (int i = 0; i < 8; ++i) z[i] = 0.0f;
    return z;
}

// ---- CDNA5 async global->LDS copy (per-lane b128), ASYNCcnt tracked ----
__device__ __forceinline__ void async_b128(void* lds, const void* gaddr) {
    unsigned off = (unsigned)(uintptr_t)lds;   // low 32b of generic = LDS offset
    asm volatile("global_load_async_to_lds_b128 %0, %1, off"
                 :: "v"(off), "v"(gaddr) : "memory");
}
__device__ __forceinline__ void async_wait0() {
    asm volatile("s_wait_asynccnt 0x0" ::: "memory");
}

// ---- fragment loaders: 16 bf16 = 32 bytes, from 16B-aligned LDS chunks ----
struct frag_bits { uint4 a, b; };
__device__ __forceinline__ v16bf frag_contig(const unsigned short* p) {
    frag_bits u;
    u.a = *(const uint4*)p;
    u.b = *(const uint4*)(p + 8);
    return __builtin_bit_cast(v16bf, u);
}
__device__ __forceinline__ v16bf frag_split(const unsigned short* p0,
                                            const unsigned short* p1) {
    frag_bits u;
    u.a = *(const uint4*)p0;
    u.b = *(const uint4*)p1;
    return __builtin_bit_cast(v16bf, u);
}

// ---- DPP16 butterfly reduction within each 16-lane half (pure VALU) ----
// XOR masks {1,2,7,15}: linearly independent -> exact butterfly for max & sum.
template <int CTRL>
__device__ __forceinline__ float dppf(float x) {
    return __builtin_bit_cast(float,
        __builtin_amdgcn_update_dpp(0, __builtin_bit_cast(int, x),
                                    CTRL, 0xF, 0xF, true));
}
__device__ __forceinline__ float rowmax16(float x) {
    x = fmaxf(x, dppf<0xB1>(x));    // quad_perm [1,0,3,2]  (xor 1)
    x = fmaxf(x, dppf<0x4E>(x));    // quad_perm [2,3,0,1]  (xor 2)
    x = fmaxf(x, dppf<0x141>(x));   // row_half_mirror      (xor 7)
    x = fmaxf(x, dppf<0x140>(x));   // row_mirror           (xor 15)
    return x;
}
__device__ __forceinline__ float rowsum16(float x) {
    x += dppf<0xB1>(x);
    x += dppf<0x4E>(x);
    x += dppf<0x141>(x);
    x += dppf<0x140>(x);
    return x;
}

// ---------------------------------------------------------------------------
// fp32 -> bf16 conversion, 4 elements/thread
// ---------------------------------------------------------------------------
__global__ void cvt_bf16_k(const float* __restrict__ in,
                           unsigned short* __restrict__ out, int n) {
    int i = (blockIdx.x * blockDim.x + threadIdx.x) * 4;
    if (i + 3 < n) {
        float4 f = *(const float4*)(in + i);
        uint2 pk;
        pk.x = (unsigned)f2bf(f.x) | ((unsigned)f2bf(f.y) << 16);
        pk.y = (unsigned)f2bf(f.z) | ((unsigned)f2bf(f.w) << 16);
        *(uint2*)(out + i) = pk;
    }
}

// ---------------------------------------------------------------------------
// fp32 [R][C] -> bf16 transposed [C][R] (weights, one-time)
// ---------------------------------------------------------------------------
__global__ __launch_bounds__(256)
void cvt_bf16_T_k(const float* __restrict__ in,
                  unsigned short* __restrict__ out, int R, int C) {
    __shared__ float tile[32][33];
    const int c0 = blockIdx.x * 32, r0 = blockIdx.y * 32;
    const int tx = threadIdx.x & 31, ty = threadIdx.x >> 5;   // 32x8
#pragma unroll
    for (int i = ty; i < 32; i += 8)
        tile[i][tx] = in[(size_t)(r0 + i) * C + c0 + tx];
    __syncthreads();
#pragma unroll
    for (int i = ty; i < 32; i += 8)
        out[(size_t)(c0 + i) * R + r0 + tx] = f2bf(tile[tx][i]);
}

// ---------------------------------------------------------------------------
// WMMA GEMM: C[MxN] = A[MxK] * Bt^T, A row-major bf16, Bt = B^T [NxK] bf16.
// 128 threads (4 waves), 128x64 tile; each wave owns 32 rows (2 groups), so
// every B-fragment feeds two WMMAs. Double-buffered LDS filled by async
// global->LDS DMA: one barrier per k-step, no staging VGPRs.
// CF32=0 -> bf16 out (scaled), CF32=1 -> fp32 out.
// ---------------------------------------------------------------------------
template <int CF32>
__global__ __launch_bounds__(128)
void gemm_bf16_k(const unsigned short* __restrict__ A,
                 const unsigned short* __restrict__ Bt,
                 void* __restrict__ Cv, int M, int N, int K, float cscale) {
    __shared__ unsigned short As[2][128][40];  // 80B rows (16B multiple)
    __shared__ unsigned short Bs[2][64][40];

    const int tid  = threadIdx.x;
    const int wave = tid >> 5;
    const int lane = tid & 31;
    const int m0 = blockIdx.x * 128;
    const int n0 = blockIdx.y * 64;

    const int blr = tid >> 1;                // B loader row (0..63)
    const int blc = (tid & 1) * 16;          // B loader col base (0/16)

    v8f acc[2][4];
#pragma unroll
    for (int g = 0; g < 2; ++g)
#pragma unroll
        for (int n = 0; n < 4; ++n) acc[g][n] = v8f_zero();

    auto issue = [&](int k0, int buf) {      // async DMA of next tile
        const unsigned short* pa = &A [(size_t)(m0 + tid) * K + k0];
        const unsigned short* pb = &Bt[(size_t)(n0 + blr) * K + k0 + blc];
#pragma unroll
        for (int j = 0; j < 4; ++j)
            async_b128(&As[buf][tid][j * 8], pa + j * 8);
        async_b128(&Bs[buf][blr][blc],     pb);
        async_b128(&Bs[buf][blr][blc + 8], pb + 8);
    };

    issue(0, 0);
    const int nk = K / 32;
    for (int kt = 0; kt < nk; ++kt) {
        const int cur = kt & 1;
        async_wait0();                        // my async stores into cur done
        __syncthreads();                      // everyone's done -> tile ready
        if (kt + 1 < nk) issue((kt + 1) * 32, cur ^ 1);

        const int koff = (lane >> 4) * 8;
        const int kb2  = (lane >> 4) * 16;
        v16bf afrag[2];
#pragma unroll
        for (int g = 0; g < 2; ++g) {
            int arow = 32 * wave + 16 * g + (lane & 15);
            afrag[g] = frag_split(&As[cur][arow][koff], &As[cur][arow][16 + koff]);
        }
#pragma unroll
        for (int n = 0; n < 4; ++n) {
            v16bf bfrag = frag_split(&Bs[cur][16 * n + (lane & 15)][koff], &Bs[cur][16 * n + (lane & 15)][16 + koff]);
#pragma unroll
            for (int g = 0; g < 2; ++g)       // B reused by both groups
                acc[g][n] = __builtin_amdgcn_wmma_f32_16x16x32_bf16(
                    false, afrag[g], false, bfrag, (short)0, acc[g][n], false, false);
        }
        // no trailing barrier: writes for step kt+2 (into cur) are issued only
        // after all waves pass the kt+1 barrier, i.e. after this compute.
    }

#pragma unroll
    for (int g = 0; g < 2; ++g)
#pragma unroll
        for (int n = 0; n < 4; ++n) {
            int col = n0 + 16 * n + (lane & 15);
#pragma unroll
            for (int r = 0; r < 8; ++r) {
                int row = m0 + 32 * wave + 16 * g + 8 * (lane >> 4) + r;
                float v = acc[g][n][r] * cscale;
                if (CF32) ((float*)Cv)[(size_t)row * N + col] = v;
                else      ((unsigned short*)Cv)[(size_t)row * N + col] = f2bf(v);
            }
        }
}

// ---------------------------------------------------------------------------
// Flash attention: block = (b,h) x 128-query tile; 4 waves x 2 groups x 16 rows.
// Q pre-scaled by log2e/sqrt(Dh) -> online softmax runs in exp2 domain.
// K chunks via async global->LDS DMA into double-buffered LDS; V transposed
// on store through registers; every K/V fragment feeds two WMMAs;
// sub-diagonal chunks take a mask-free path. One block barrier per chunk
// (P tiles are wave-private; same-wave LDS ops are processed in order).
// ---------------------------------------------------------------------------
__global__ __launch_bounds__(128)
void attn_k(const unsigned short* __restrict__ Q,
            const unsigned short* __restrict__ Kg,
            const unsigned short* __restrict__ Vg,
            unsigned short* __restrict__ O) {
    __shared__ unsigned short Qs[128][72];        // 144B rows
    __shared__ unsigned short Ks[2][32][72];      // key-major (d contiguous)
    __shared__ unsigned short VsT[2][64][40];     // d-major   (key contiguous)
    __shared__ unsigned short Ps[4][2][16][40];   // per-wave, per-group P tile

    const int tid  = threadIdx.x;
    const int wave = tid >> 5;
    const int lane = tid & 31;
    const int bh = blockIdx.x;
    const int b  = bh >> 4;
    const int h  = bh & 15;
    const int q0 = blockIdx.y * 128;
    const size_t rowbase = (size_t)b * SEQ * DMODEL + h * DHEAD;

    {   // stage 128x64 Q tile (one row per thread, 8x b128)
        const unsigned short* src = &Q[rowbase + (size_t)(q0 + tid) * DMODEL];
#pragma unroll
        for (int j = 0; j < 8; ++j)
            *(uint4*)&Qs[tid][j * 8] = ((const uint4*)src)[j];
    }
    __syncthreads();

    const int koff = (lane >> 4) * 8;
    const int kb2  = (lane >> 4) * 16;

    // Q fragments: 2 groups x 2 k-steps, kept in registers for whole kernel
    v16bf qa[2][2];
#pragma unroll
    for (int g = 0; g < 2; ++g) {
        int row = 64 * g + 16 * wave + (lane & 15);
#pragma unroll
        for (int s = 0; s < 2; ++s)
            qa[g][s] = frag_split(&Qs[row][32 * s + koff],
                                  &Qs[row][32 * s + 16 + koff]);
    }

    v8f oacc[2][4];
#pragma unroll
    for (int g = 0; g < 2; ++g)
#pragma unroll
        for (int n = 0; n < 4; ++n) oacc[g][n] = v8f_zero();
    float mrow[2][8], lrow[2][8];
#pragma unroll
    for (int g = 0; g < 2; ++g)
#pragma unroll
        for (int r = 0; r < 8; ++r) { mrow[g][r] = -1e30f; lrow[g][r] = 0.0f; }

    const int kr  = tid >> 2;            // K/V loader row (0..31)
    const int kcb = (tid & 3) * 16;      // loader col base
    uint4 vreg[2];
    auto issueK = [&](int kbase, int buf) {
        const unsigned short* ks = &Kg[rowbase + (size_t)(kbase + kr) * DMODEL + kcb];
        async_b128(&Ks[buf][kr][kcb],     ks);
        async_b128(&Ks[buf][kr][kcb + 8], ks + 8);
    };
    auto fetchV = [&](int kbase) {
        const unsigned short* vs = &Vg[rowbase + (size_t)(kbase + kr) * DMODEL + kcb];
        vreg[0] = ((const uint4*)vs)[0]; vreg[1] = ((const uint4*)vs)[1];
    };

    // per-chunk compute: S' = Q*K^T (exp2 domain), online softmax, O += P*V
    auto process = [&](int kbase, int cur, auto maskc) {
        constexpr bool MASK = decltype(maskc)::value;
        v8f sacc[2][2];                       // [group][key sub-tile]
#pragma unroll
        for (int g = 0; g < 2; ++g)
#pragma unroll
            for (int j = 0; j < 2; ++j) sacc[g][j] = v8f_zero();
#pragma unroll
        for (int j = 0; j < 2; ++j) {
            int key = 16 * j + (lane & 15);
#pragma unroll
            for (int s = 0; s < 2; ++s) {
                v16bf kf = frag_split(&Ks[cur][key][32 * s + koff], &Ks[cur][key][32 * s + 16 + koff]);
#pragma unroll
                for (int g = 0; g < 2; ++g)   // K fragment reused by both groups
                    sacc[g][j] = __builtin_amdgcn_wmma_f32_16x16x32_bf16(
                        false, qa[g][s], false, kf, (short)0, sacc[g][j], false, false);
            }
        }
#pragma unroll
        for (int g = 0; g < 2; ++g) {
            const int keyc  = kbase + (lane & 15);
            const int qrowb = q0 + 64 * g + 16 * wave + 8 * (lane >> 4);
            const int rowoff = 8 * (lane >> 4);
            float fac[8];
#pragma unroll
            for (int r = 0; r < 8; ++r) {
                float s0 = sacc[g][0][r];
                float s1 = sacc[g][1][r];
                if (MASK) {
                    int q = qrowb + r;
                    if (keyc      > q) s0 = -1e30f;
                    if (keyc + 16 > q) s1 = -1e30f;
                }
                float mnew = fmaxf(mrow[g][r], rowmax16(fmaxf(s0, s1)));
                float f  = exp2f(mrow[g][r] - mnew);     // bare v_exp_f32
                float e0 = exp2f(s0 - mnew);
                float e1 = exp2f(s1 - mnew);
                lrow[g][r] = lrow[g][r] * f + rowsum16(e0 + e1);
                mrow[g][r] = mnew;
                fac[r] = f;
                Ps[wave][g][rowoff + r][(lane & 15)]      = f2bf(e0);
                Ps[wave][g][rowoff + r][16 + (lane & 15)] = f2bf(e1);
            }
#pragma unroll
            for (int n = 0; n < 4; ++n)
#pragma unroll
                for (int r = 0; r < 8; ++r) oacc[g][n][r] *= fac[r];
        }
        // Ps is wave-private: same-wave LDS ops are in order; just keep the
        // scheduler from hoisting the reads above the writes.
        __builtin_amdgcn_wave_barrier();

        v16bf pa[2];
#pragma unroll
        for (int g = 0; g < 2; ++g)
            pa[g] = frag_split(&Ps[wave][g][lane & 15][koff],
                               &Ps[wave][g][lane & 15][16 + koff]);
#pragma unroll
        for (int n = 0; n < 4; ++n) {
            v16bf vf = frag_split(&VsT[cur][16 * n + (lane & 15)][koff], &VsT[cur][16 * n + (lane & 15)][16 + koff]);
#pragma unroll
            for (int g = 0; g < 2; ++g)       // V fragment reused by both groups
                oacc[g][n] = __builtin_amdgcn_wmma_f32_16x16x32_bf16(
                    false, pa[g], false, vf, (short)0, oacc[g][n], false, false);
        }
        // no trailing barrier: next chunk writes the other K/V buffer, and its
        // Ps writes are wave-local.
    };

    const int nchunk = (q0 + 128) >> 5;  // causal coverage for this tile
    const int nfull  = q0 >> 5;          // chunks fully below the diagonal
    issueK(0, 0);
    fetchV(0);
    for (int kc = 0; kc < nchunk; ++kc) {
        const int kbase = kc * 32;
        const int cur = kc & 1;
        // V transpose store into current buffer (from regs fetched last iter)
#pragma unroll
        for (int qq = 0; qq < 2; ++qq) {
            unsigned int wd[4] = { vreg[qq].x, vreg[qq].y, vreg[qq].z, vreg[qq].w };
#pragma unroll
            for (int d = 0; d < 4; ++d) {
                VsT[cur][kcb + qq * 8 + 2 * d    ][kr] = (unsigned short)(wd[d] & 0xFFFFu);
                VsT[cur][kcb + qq * 8 + 2 * d + 1][kr] = (unsigned short)(wd[d] >> 16);
            }
        }
        async_wait0();                       // my K DMA into cur complete
        __syncthreads();                     // all staging visible block-wide
        if (kc + 1 < nchunk) { issueK(kbase + 32, cur ^ 1); fetchV(kbase + 32); }

        if (kc < nfull) process(kbase, cur, BoolC<false>{});   // mask-free bulk
        else            process(kbase, cur, BoolC<true>{});    // diagonal
    }

    // normalize + store bf16
#pragma unroll
    for (int g = 0; g < 2; ++g) {
        float inv[8];
#pragma unroll
        for (int r = 0; r < 8; ++r)
            inv[r] = (lrow[g][r] > 0.0f) ? 1.0f / lrow[g][r] : 0.0f;
#pragma unroll
        for (int n = 0; n < 4; ++n) {
            int col = 16 * n + (lane & 15);
#pragma unroll
            for (int r = 0; r < 8; ++r) {
                int row = q0 + 64 * g + 16 * wave + 8 * (lane >> 4) + r;
                O[rowbase + (size_t)row * DMODEL + col] = f2bf(oacc[g][n][r] * inv[r]);
            }
        }
    }
}

// ---------------------------------------------------------------------------
// Launcher
// ---------------------------------------------------------------------------
extern "C" void kernel_launch(void* const* d_in, const int* in_sizes, int n_in,
                              void* d_out, int out_size, void* d_ws, size_t ws_size,
                              hipStream_t stream) {
    const float* x  = (const float*)d_in[0];
    const float* WQ = (const float*)d_in[1];
    const float* WK = (const float*)d_in[2];
    const float* WV = (const float*)d_in[3];
    const float* WO = (const float*)d_in[4];

    const size_t nX = (size_t)MTOT * DMODEL;     // 8M
    const size_t nW = (size_t)DMODEL * DMODEL;   // 1M

    char* w = (char*)d_ws;
    unsigned short* xb  = (unsigned short*)w; w += nX * 2;
    unsigned short* wqt = (unsigned short*)w; w += nW * 2;   // W^T layouts
    unsigned short* wkt = (unsigned short*)w; w += nW * 2;
    unsigned short* wvt = (unsigned short*)w; w += nW * 2;
    unsigned short* wot = (unsigned short*)w; w += nW * 2;
    unsigned short* Qb  = (unsigned short*)w; w += nX * 2;
    unsigned short* Kb  = (unsigned short*)w; w += nX * 2;
    unsigned short* Vb  = (unsigned short*)w; w += nX * 2;
    unsigned short* Ob  = (unsigned short*)w; w += nX * 2;

    // 1) fp32 -> bf16 (x), fp32 -> bf16 transposed (weights)
    cvt_bf16_k<<<(int)((nX / 4 + 255) / 256), 256, 0, stream>>>(x, xb, (int)nX);
    dim3 gt(DMODEL / 32, DMODEL / 32);
    cvt_bf16_T_k<<<gt, 256, 0, stream>>>(WQ, wqt, DMODEL, DMODEL);
    cvt_bf16_T_k<<<gt, 256, 0, stream>>>(WK, wkt, DMODEL, DMODEL);
    cvt_bf16_T_k<<<gt, 256, 0, stream>>>(WV, wvt, DMODEL, DMODEL);
    cvt_bf16_T_k<<<gt, 256, 0, stream>>>(WO, wot, DMODEL, DMODEL);

    // 2) Q/K/V projections (Q gets log2e/sqrt(Dh) folded into epilogue)
    dim3 gg(MTOT / 128, DMODEL / 64);
    gemm_bf16_k<0><<<gg, 128, 0, stream>>>(xb, wqt, Qb, MTOT, DMODEL, DMODEL, QSCALE);
    gemm_bf16_k<0><<<gg, 128, 0, stream>>>(xb, wkt, Kb, MTOT, DMODEL, DMODEL, 1.0f);
    gemm_bf16_k<0><<<gg, 128, 0, stream>>>(xb, wvt, Vb, MTOT, DMODEL, DMODEL, 1.0f);

    // 3) flash attention (128-row q tiles)
    attn_k<<<dim3(BATCH * NHEADS, SEQ / 128), 128, 0, stream>>>(Qb, Kb, Vb, Ob);

    // 4) output projection (fp32 epilogue)
    gemm_bf16_k<1><<<gg, 128, 0, stream>>>(Ob, wot, d_out, MTOT, DMODEL, DMODEL, 1.0f);
}
